// Model_29592324669622
// MI455X (gfx1250) — compile-verified, matrix operand order checked by reading, not run on hardware
//
#include <hip/hip_runtime.h>
#include <hip/hip_bf16.h>

// ---------------------------------------------------------------------------
// Problem constants (match reference)
// ---------------------------------------------------------------------------
#define NU 100000
#define NM 20000
#define HD 64
#define NE 1000000
#define NEL 500000
#define NL 3
#define BN_EPS 1e-5f

typedef __attribute__((ext_vector_type(16))) __bf16 bf16x16;
typedef __attribute__((ext_vector_type(8)))  float  f32x8;

__device__ __forceinline__ __bf16 to_bf16(float f) { return (__bf16)f; }

// Build one 16x32 bf16 A fragment for this lane.
// p = lane's A row base (+ k0).  Layout (ISA 7.12.2): element j holds
// K = (j/8)*16 + half*8 + (j%8)  ->  four contiguous float4 runs.
__device__ __forceinline__ bf16x16 load_a_frag(const float* __restrict__ p, int half)
{
    const float4 a0 = *(const float4*)(p + half * 8);
    const float4 a1 = *(const float4*)(p + half * 8 + 4);
    const float4 a2 = *(const float4*)(p + 16 + half * 8);
    const float4 a3 = *(const float4*)(p + 16 + half * 8 + 4);
    bf16x16 r;
    r[0]  = to_bf16(a0.x); r[1]  = to_bf16(a0.y); r[2]  = to_bf16(a0.z); r[3]  = to_bf16(a0.w);
    r[4]  = to_bf16(a1.x); r[5]  = to_bf16(a1.y); r[6]  = to_bf16(a1.z); r[7]  = to_bf16(a1.w);
    r[8]  = to_bf16(a2.x); r[9]  = to_bf16(a2.y); r[10] = to_bf16(a2.z); r[11] = to_bf16(a2.w);
    r[12] = to_bf16(a3.x); r[13] = to_bf16(a3.y); r[14] = to_bf16(a3.z); r[15] = to_bf16(a3.w);
    return r;
}

// ---------------------------------------------------------------------------
// Templated WMMA GEMM:
//   C[:, offC:offC+N] = act( A0[idx0][:, off0:] @ W0 (+ A1[idx1][:, off1:] @ W1) + bias )
// - W0/W1 staged global->LDS with GLOBAL_LOAD_ASYNC_TO_LDS_B128 (ASYNCcnt).
// - Each wave owns one n-tile; its B fragments are built once into registers
//   and reused across a grid-strided m-tile loop.
// - Block = 128 threads = 4 waves.  M, N multiples of 16; K multiples of 32.
// ---------------------------------------------------------------------------
template <int K0, int K1, int N, bool RELU>
__global__ void gemm_wmma(const float* __restrict__ A0, const int* __restrict__ idx0,
                          int lda0, int off0, const float* __restrict__ W0,
                          const float* __restrict__ A1, const int* __restrict__ idx1,
                          int lda1, int off1, const float* __restrict__ W1,
                          const float* __restrict__ bias,
                          float* __restrict__ C, int ldc, int offC, int M)
{
    constexpr int NKB0 = K0 / 32;
    constexpr int NKB1 = K1 / 32;
    constexpr int NKB  = NKB0 + NKB1;
    constexpr int NTN  = N / 16;
    constexpr int MW   = 4 / NTN;          // m-subgroups per 4-wave block

    extern __shared__ float smemW[];       // [K0*N] then [K1*N] floats

    const int tid = threadIdx.x;

    // ---- async stage W0 (and W1) into LDS: 16B per lane per issue ----
    {
        const unsigned ldsBase = (unsigned)(uintptr_t)smemW;   // low 32b = LDS byte addr
        for (int i = tid * 4; i < K0 * N; i += 128 * 4) {
            unsigned           lds = ldsBase + (unsigned)(i * 4);
            unsigned long long ga  = (unsigned long long)(const void*)(W0 + i);
            asm volatile("global_load_async_to_lds_b128 %0, %1, off"
                         :: "v"(lds), "v"(ga) : "memory");
        }
        if constexpr (K1 > 0) {
            const unsigned ldsB1 = ldsBase + (unsigned)(K0 * N * 4);
            for (int i = tid * 4; i < K1 * N; i += 128 * 4) {
                unsigned           lds = ldsB1 + (unsigned)(i * 4);
                unsigned long long ga  = (unsigned long long)(const void*)(W1 + i);
                asm volatile("global_load_async_to_lds_b128 %0, %1, off"
                             :: "v"(lds), "v"(ga) : "memory");
            }
        }
        asm volatile("s_wait_asynccnt 0x0" ::: "memory");
        __syncthreads();
    }

    const int lane  = tid & 31;
    const int wave  = tid >> 5;
    const int r16   = lane & 15;
    const int half  = lane >> 4;
    const int nt    = wave % NTN;          // this wave's fixed n-tile
    const int mgrp  = wave / NTN;          // m-subgroup within block
    const int ncol  = nt * 16 + r16;

    // ---- build B fragments once (registers), from LDS ----
    bf16x16 bfr[NKB];
#pragma unroll
    for (int kb = 0; kb < NKB; ++kb) {
        const float* Wb = (kb < NKB0) ? smemW : (smemW + K0 * N);
        const int klo   = (kb < NKB0) ? kb * 32 : (kb - NKB0) * 32;
#pragma unroll
        for (int j = 0; j < 16; ++j) {
            const int k = klo + ((j >> 3) << 4) + (half << 3) + (j & 7);
            bfr[kb][j] = to_bf16(Wb[(size_t)k * N + ncol]);
        }
    }

    // ---- grid-strided m-tile loop (B fragments reused every iteration) ----
    const int mtiles = M >> 4;
    for (int mt = blockIdx.x * MW + mgrp; mt < mtiles; mt += gridDim.x * MW) {
        const int mlog = mt * 16 + r16;    // lane's logical A row
        f32x8 acc = {};
        {
            const int ar = idx0 ? idx0[mlog] : mlog;
            const float* Ar = A0 + (size_t)ar * lda0 + off0;
#pragma unroll
            for (int kb = 0; kb < NKB0; ++kb) {
                bf16x16 a = load_a_frag(Ar + kb * 32, half);
                acc = __builtin_amdgcn_wmma_f32_16x16x32_bf16(
                    false, a, false, bfr[kb], (short)0, acc, false, false);
            }
        }
        if constexpr (K1 > 0) {
            const int ar = idx1 ? idx1[mlog] : mlog;
            const float* Ar = A1 + (size_t)ar * lda1 + off1;
#pragma unroll
            for (int kb = 0; kb < NKB1; ++kb) {
                bf16x16 a = load_a_frag(Ar + kb * 32, half);
                acc = __builtin_amdgcn_wmma_f32_16x16x32_bf16(
                    false, a, false, bfr[NKB0 + kb], (short)0, acc, false, false);
            }
        }
        const float bv = bias ? bias[ncol] : 0.0f;
#pragma unroll
        for (int r = 0; r < 8; ++r) {      // C layout: VGPR r -> M = r + 8*half
            const int mrow = mt * 16 + half * 8 + r;
            float v = acc[r] + bv;
            if (RELU) v = fmaxf(v, 0.0f);
            C[(size_t)mrow * ldc + offC + ncol] = v;
        }
    }
}

// ---------------------------------------------------------------------------
// Edge scatter-add (mean aggregation).  One thread per (edge, 4-col group):
// float4 gather + 4 global f32 atomics; count once per edge.
// ---------------------------------------------------------------------------
__global__ void scatter_accum(const float* __restrict__ X, const int* __restrict__ xidx,
                              int ldx, int xoff,
                              const int* __restrict__ esrc, const int* __restrict__ edst,
                              int E, float* __restrict__ agg, float* __restrict__ cnt)
{
    int gid = blockIdx.x * blockDim.x + threadIdx.x;
    if (gid >= E * (HD / 4)) return;
    int e = gid >> 4, q = gid & 15;
    int s = esrc[e], d = edst[e];
    int sr = xidx ? xidx[s] : s;
    const float4 v = *(const float4*)(X + (size_t)sr * ldx + xoff + q * 4);
    float* dst = agg + (size_t)d * HD + q * 4;
    atomicAdd(dst + 0, v.x);
    atomicAdd(dst + 1, v.y);
    atomicAdd(dst + 2, v.z);
    atomicAdd(dst + 3, v.w);
    if (q == 0) atomicAdd(&cnt[d], 1.0f);
}

__global__ void mean_div(float* __restrict__ agg, const float* __restrict__ cnt, int n)
{
    int gid = blockIdx.x * blockDim.x + threadIdx.x;
    if (gid < n) agg[gid] /= fmaxf(cnt[gid >> 6], 1.0f);
}

// ---------------------------------------------------------------------------
// BatchNorm (training-mode, biased variance)
// ---------------------------------------------------------------------------
__global__ void bn_stats(const float* __restrict__ X, int M, int N,
                         float* __restrict__ sum, float* __restrict__ sumsq)
{
    int c = threadIdx.x;                   // blockDim.x == N (32 or 64)
    float s = 0.0f, q = 0.0f;
    for (int r = blockIdx.x; r < M; r += gridDim.x) {
        float v = X[(size_t)r * N + c];
        s += v; q += v * v;
    }
    atomicAdd(&sum[c], s);
    atomicAdd(&sumsq[c], q);
}

__global__ void bn_apply_relu(float* __restrict__ X, int M, int N,
                              const float* __restrict__ sum, const float* __restrict__ sumsq,
                              const float* __restrict__ gamma, const float* __restrict__ beta)
{
    int gid = blockIdx.x * blockDim.x + threadIdx.x;
    if (gid >= M * N) return;
    int c = gid % N;
    float invM = 1.0f / (float)M;
    float m = sum[c] * invM;
    float var = sumsq[c] * invM - m * m;
    float v = gamma[c] * (X[gid] - m) * rsqrtf(var + BN_EPS) + beta[c];
    X[gid] = fmaxf(v, 0.0f);
}

// Final 32 -> 1 projection (too thin for WMMA)
__global__ void final_proj(const float* __restrict__ Z, const float* __restrict__ W3,
                           const float* __restrict__ b3, float* __restrict__ out, int M)
{
    int r = blockIdx.x * blockDim.x + threadIdx.x;
    if (r >= M) return;
    float s = b3[0];
#pragma unroll
    for (int k = 0; k < 32; ++k) s += Z[(size_t)r * 32 + k] * W3[k];
    out[r] = s;
}

// ---------------------------------------------------------------------------
// Host-side orchestration
// ---------------------------------------------------------------------------
static inline int gemm_blocks(int mtiles, int mw)
{
    int need = (mtiles + mw - 1) / mw;
    return need < 2048 ? need : 2048;
}

extern "C" void kernel_launch(void* const* d_in, const int* in_sizes, int n_in,
                              void* d_out, int out_size, void* d_ws, size_t ws_size,
                              hipStream_t stream)
{
    (void)in_sizes; (void)n_in; (void)out_size; (void)ws_size;

    const int*   n_id        = (const int*)  d_in[0];
    const float* movie_x     = (const float*)d_in[1];
    const int*   edge_user   = (const int*)  d_in[2];
    const int*   edge_movie  = (const int*)  d_in[3];
    const int*   label_user  = (const int*)  d_in[4];
    const int*   label_movie = (const int*)  d_in[5];
    const float* user_emb    = (const float*)d_in[6];
    const float* Wl_um       = (const float*)d_in[7];
    const float* b_um        = (const float*)d_in[8];
    const float* Wr_um       = (const float*)d_in[9];
    const float* Wl_mu       = (const float*)d_in[10];
    const float* b_mu        = (const float*)d_in[11];
    const float* Wr_mu       = (const float*)d_in[12];
    const float* proj_u_W    = (const float*)d_in[13];
    const float* proj_u_b    = (const float*)d_in[14];
    const float* proj_m_W    = (const float*)d_in[15];
    const float* proj_m_b    = (const float*)d_in[16];
    const float* W1          = (const float*)d_in[17];
    const float* b1          = (const float*)d_in[18];
    const float* g1          = (const float*)d_in[19];
    const float* be1         = (const float*)d_in[20];
    const float* W2          = (const float*)d_in[21];
    const float* b2          = (const float*)d_in[22];
    const float* g2          = (const float*)d_in[23];
    const float* be2         = (const float*)d_in[24];
    const float* W3          = (const float*)d_in[25];
    const float* b3          = (const float*)d_in[26];
    float*       out         = (float*)d_out;

    // ---- workspace layout (floats) ----
    float* h_u   = (float*)d_ws;                       // [NU, 64]
    float* h_m   = h_u + (size_t)NU * HD;              // [NM, 64]
    float* stats = h_m + (size_t)NM * HD;              // 256 floats
    float* sum1  = stats;        float* sq1 = stats + 64;
    float* sum2  = stats + 128;  float* sq2 = stats + 160;
    float* uni   = stats + 256;
    // layout A (graph phase) — dead before layout B is written
    float* agg_u  = uni;                               // [NU, 64]
    float* agg_m  = agg_u + (size_t)NU * HD;           // [NM, 64]
    float* cnt_u  = agg_m + (size_t)NM * HD;           // [NU]
    float* cnt_m  = cnt_u + NU;                        // [NM]
    float* outs_u = cnt_m + NM;                        // [NU, 192] (JK concat)
    float* outs_m = outs_u + (size_t)NU * 3 * HD;      // [NM, 192]
    // layout B (classifier phase) overlaps layout A
    float* z1 = uni;                                   // [EL, 64]
    float* z2 = z1 + (size_t)NEL * HD;                 // [EL, 32]

    const size_t aggBytes = ((size_t)NU * HD + (size_t)NM * HD + NU + NM) * sizeof(float);

    dim3 blk256(256), blkG(128);
    const int scatterBlocks = (NE * (HD / 4) + 255) / 256;
    const size_t shm1 = (size_t)(HD + HD) * HD * 4;          // dual 64+64, N=64: 32KB
    const size_t shmP = (size_t)(3 * HD) * HD * 4;           // K=192, N=64: 48KB
    const size_t shm2 = (size_t)HD * (HD / 2) * 4;           // K=64, N=32: 8KB

    // ---------------- 3 SAGE layers ----------------
    for (int i = 0; i < NL; ++i) {
        const float* xu  = (i == 0) ? user_emb : outs_u;
        const int*   xui = (i == 0) ? n_id : nullptr;
        int          ldu = (i == 0) ? HD : 3 * HD;
        int          ofu = (i == 0) ? 0  : (i - 1) * HD;
        const float* xm  = (i == 0) ? movie_x : outs_m;
        int          ldm = (i == 0) ? HD : 3 * HD;
        int          ofm = (i == 0) ? 0  : (i - 1) * HD;

        hipMemsetAsync(agg_u, 0, aggBytes, stream);

        scatter_accum<<<scatterBlocks, blk256, 0, stream>>>(
            xu, xui, ldu, ofu, edge_user, edge_movie, NE, agg_m, cnt_m);
        scatter_accum<<<scatterBlocks, blk256, 0, stream>>>(
            xm, nullptr, ldm, ofm, edge_movie, edge_user, NE, agg_u, cnt_u);

        mean_div<<<(NM * HD + 255) / 256, blk256, 0, stream>>>(agg_m, cnt_m, NM * HD);
        mean_div<<<(NU * HD + 255) / 256, blk256, 0, stream>>>(agg_u, cnt_u, NU * HD);

        // new_m = relu(agg_m @ Wl_um[i] + x_m @ Wr_um[i] + b_um[i]) -> outs_m[:, i*64]
        gemm_wmma<HD, HD, HD, true><<<gemm_blocks(NM / 16, 1), blkG, shm1, stream>>>(
            agg_m, nullptr, HD, 0, Wl_um + (size_t)i * HD * HD,
            xm, nullptr, ldm, ofm, Wr_um + (size_t)i * HD * HD,
            b_um + (size_t)i * HD, outs_m, 3 * HD, i * HD, NM);
        // new_u = relu(agg_u @ Wl_mu[i] + x_u @ Wr_mu[i] + b_mu[i]) -> outs_u[:, i*64]
        gemm_wmma<HD, HD, HD, true><<<gemm_blocks(NU / 16, 1), blkG, shm1, stream>>>(
            agg_u, nullptr, HD, 0, Wl_mu + (size_t)i * HD * HD,
            xu, xui, ldu, ofu, Wr_mu + (size_t)i * HD * HD,
            b_mu + (size_t)i * HD, outs_u, 3 * HD, i * HD, NU);
    }

    // ---------------- JumpingKnowledge-cat projections ----------------
    gemm_wmma<3 * HD, 0, HD, false><<<gemm_blocks(NU / 16, 1), blkG, shmP, stream>>>(
        outs_u, nullptr, 3 * HD, 0, proj_u_W,
        nullptr, nullptr, 0, 0, nullptr,
        proj_u_b, h_u, HD, 0, NU);
    gemm_wmma<3 * HD, 0, HD, false><<<gemm_blocks(NM / 16, 1), blkG, shmP, stream>>>(
        outs_m, nullptr, 3 * HD, 0, proj_m_W,
        nullptr, nullptr, 0, 0, nullptr,
        proj_m_b, h_m, HD, 0, NM);

    // ---------------- edge classifier ----------------
    // z1 = [h_u[label_user] | h_m[label_movie]] @ W1 + b1  (gather fused into GEMM)
    gemm_wmma<HD, HD, HD, false><<<gemm_blocks(NEL / 16, 1), blkG, shm1, stream>>>(
        h_u, label_user, HD, 0, W1,
        h_m, label_movie, HD, 0, W1 + (size_t)HD * HD,
        b1, z1, HD, 0, NEL);

    hipMemsetAsync(stats, 0, 256 * sizeof(float), stream);
    bn_stats<<<2048, dim3(HD), 0, stream>>>(z1, NEL, HD, sum1, sq1);
    bn_apply_relu<<<(NEL * HD + 255) / 256, blk256, 0, stream>>>(z1, NEL, HD, sum1, sq1, g1, be1);

    // z2 = z1 @ W2 + b2
    gemm_wmma<HD, 0, HD / 2, false><<<gemm_blocks(NEL / 16, 2), blkG, shm2, stream>>>(
        z1, nullptr, HD, 0, W2,
        nullptr, nullptr, 0, 0, nullptr,
        b2, z2, HD / 2, 0, NEL);

    bn_stats<<<2048, dim3(HD / 2), 0, stream>>>(z2, NEL, HD / 2, sum2, sq2);
    bn_apply_relu<<<(NEL * HD / 2 + 255) / 256, blk256, 0, stream>>>(z2, NEL, HD / 2, sum2, sq2, g2, be2);

    // out = z2 @ W3 + b3
    final_proj<<<(NEL + 255) / 256, blk256, 0, stream>>>(z2, W3, b3, out, NEL);
}
